// MHSA_53584011984933
// MI455X (gfx1250) — matrix operand order repaired and checked by reading, not compiled, not run on hardware
//
#include <hip/hip_runtime.h>
#include <stdint.h>

#define DIM   512
#define HEADS 8
#define HD    64
#define SEQ   2048
#define BATCH 4
#define QK_SCALE 0.125f   // 64^-0.5

typedef __attribute__((ext_vector_type(16))) __bf16 v16bf;
typedef __attribute__((ext_vector_type(8)))  float  v8f;

union Frag {
    v16bf v;
    __bf16 e[16];
    uint4 q[2];
};

// ---------------------------------------------------------------------------
// Kernel 1: QKV projection.  X[8192,512] fp32 @ W[512,1536] fp32 -> bf16
// Q/K/V in [B,H,N,D].  Block = 8 row-tiles x 1 col-tile; W^T tile staged in
// LDS once (shared by all 8 waves), B-fragments become ds_load_b128 pairs.
// ---------------------------------------------------------------------------
__global__ __launch_bounds__(256) void qkv_kernel(
    const float* __restrict__ X, const float* __restrict__ W,
    __bf16* __restrict__ Q, __bf16* __restrict__ K, __bf16* __restrict__ V)
{
    __shared__ __align__(16) __bf16 WT[16][520];   // [col][k], padded rows

    const int lane = threadIdx.x & 31;
    const int wid  = threadIdx.x >> 5;
    const int l16  = lane & 15;
    const bool hi  = lane >= 16;

    const int blk = blockIdx.x;        // 64 row-groups * 96 col-tiles = 6144
    const int mG  = blk / 96;          // 128-row group
    const int nT  = blk % 96;          // 16-col tile

    // Stage W^T tile: cols nT*16..+15, all 512 k rows, converted to bf16.
    for (int i = threadIdx.x; i < 16 * 512; i += 256) {
        const int k = i >> 4, c = i & 15;
        WT[c][k] = (__bf16)W[(size_t)k * (3 * DIM) + nT * 16 + c];
    }
    __syncthreads();

    const int mT = mG * 8 + wid;
    const float* xrow = X + (size_t)(mT * 16 + l16) * DIM;   // A row per lane
    const int grp = hi ? 16 : 0;

    v8f c = {};
    #pragma unroll 4
    for (int kk = 0; kk < DIM; kk += 32) {
        Frag a, b;
        // A (16x32 bf16): lane<16 holds K {0..15}, lane>=16 {16..31}
        const int aoff = kk + (hi ? 16 : 0);
        const float4 x0 = *(const float4*)(xrow + aoff);
        const float4 x1 = *(const float4*)(xrow + aoff + 4);
        const float4 x2 = *(const float4*)(xrow + aoff + 8);
        const float4 x3 = *(const float4*)(xrow + aoff + 12);
        a.e[0]=(__bf16)x0.x;  a.e[1]=(__bf16)x0.y;  a.e[2]=(__bf16)x0.z;  a.e[3]=(__bf16)x0.w;
        a.e[4]=(__bf16)x1.x;  a.e[5]=(__bf16)x1.y;  a.e[6]=(__bf16)x1.z;  a.e[7]=(__bf16)x1.w;
        a.e[8]=(__bf16)x2.x;  a.e[9]=(__bf16)x2.y;  a.e[10]=(__bf16)x2.z; a.e[11]=(__bf16)x2.w;
        a.e[12]=(__bf16)x3.x; a.e[13]=(__bf16)x3.y; a.e[14]=(__bf16)x3.z; a.e[15]=(__bf16)x3.w;
        // B (32x16): lane col = l16, K rows kk+grp..+15, contiguous in WT row.
        const __bf16* wp = &WT[l16][kk + grp];
        b.q[0] = *(const uint4*)wp;
        b.q[1] = *(const uint4*)(wp + 8);
        c = __builtin_amdgcn_wmma_f32_16x16x32_bf16(false, a.v, false, b.v,
                                                    (short)0, c, false, false);
    }

    // D layout: VGPR r -> row r + (hi?8:0), col = l16.
    const int wcol  = nT * 16 + l16;
    const int which = wcol >> 9;
    const int h     = (wcol >> 6) & 7;
    const int d     = wcol & 63;
    __bf16* dst = (which == 0) ? Q : ((which == 1) ? K : V);
    const float s = (which == 0) ? QK_SCALE : 1.0f;
    #pragma unroll
    for (int r = 0; r < 8; ++r) {
        const int m  = mT * 16 + r + (hi ? 8 : 0);
        const int b_ = m >> 11, n = m & (SEQ - 1);
        dst[(((size_t)b_ * HEADS + h) * SEQ + n) * HD + d] = (__bf16)(c[r] * s);
    }
}

// ---------------------------------------------------------------------------
// Kernel 2: flash attention.  One block = one (b,h) x 128 query rows
// (8 waves x 16 rows).  K chunk moved with global_load_async_to_lds_b128;
// V transposed into LDS through registers.  Per 32-key chunk: 4 WMMAs for S,
// online softmax (width-16 shuffles), P->LDS, 4 WMMAs for P@V.
// ---------------------------------------------------------------------------
__global__ __launch_bounds__(256) void attn_kernel(
    const __bf16* __restrict__ Q, const __bf16* __restrict__ K,
    const __bf16* __restrict__ V, float* __restrict__ Out)
{
    __shared__ __align__(16) __bf16 Klds[32 * 64];    // [key][d] row-major
    __shared__ __align__(16) __bf16 VTlds[64 * 32];   // [d][key]
    __shared__ __align__(16) __bf16 Plds[8][16 * 32]; // per-wave P, [m][key]

    const int lane = threadIdx.x & 31;
    const int wid  = threadIdx.x >> 5;
    const int l16  = lane & 15;
    const bool hi  = lane >= 16;
    const int grp  = hi ? 16 : 0;

    const int bid = blockIdx.x;          // b(2) | h(3) | qtile(4)
    const int b   = bid >> 7;
    const int h   = (bid >> 4) & 7;
    const int qt  = bid & 15;
    const int q0  = qt * 128 + wid * 16;

    const size_t headOff = ((size_t)b * HEADS + h) * SEQ * HD;
    const __bf16* Qh = Q + headOff;
    const __bf16* Kh = K + headOff;
    const __bf16* Vh = V + headOff;

    // Q strip (16 rows x 64) as two A fragments, registers for all chunks.
    Frag aq0, aq1;
    {
        const __bf16* qrow = Qh + (size_t)(q0 + l16) * HD;
        const int off = hi ? 16 : 0;
        aq0.q[0] = *(const uint4*)(qrow + off);
        aq0.q[1] = *(const uint4*)(qrow + off + 8);
        aq1.q[0] = *(const uint4*)(qrow + 32 + off);
        aq1.q[1] = *(const uint4*)(qrow + 32 + off + 8);
    }

    v8f acc0 = {}, acc1 = {}, acc2 = {}, acc3 = {};
    float mi[8], li[8];
    #pragma unroll
    for (int r = 0; r < 8; ++r) { mi[r] = -INFINITY; li[r] = 0.0f; }

    for (int kv0 = 0; kv0 < SEQ; kv0 += 32) {
        __syncthreads();   // previous chunk's LDS reads done
        {
            const int t = threadIdx.x;                     // 256 threads x 8 elems
            // K chunk: async DMA straight into LDS (one b128 per thread).
            {
                const __bf16* gK = Kh + (size_t)kv0 * HD + t * 8;
                const unsigned int ldsK = (unsigned int)(uintptr_t)(Klds + t * 8);
                asm volatile("global_load_async_to_lds_b128 %0, %1, off"
                             :: "v"(ldsK), "v"(gK) : "memory");
            }
            // V chunk: through registers, stored transposed.
            uint4 vv = *(const uint4*)(Vh + (size_t)kv0 * HD + t * 8);
            const __bf16* vs = (const __bf16*)&vv;
            const int e0 = t * 8, kq = e0 >> 6, d0 = e0 & 63;
            #pragma unroll
            for (int i = 0; i < 8; ++i) VTlds[(d0 + i) * 32 + kq] = vs[i];
            if (kv0 + 32 < SEQ) {
                __builtin_prefetch(Vh + (size_t)(kv0 + 32) * HD + t * 8, 0, 0);
                __builtin_prefetch(Kh + (size_t)(kv0 + 32) * HD + t * 8, 0, 0);
            }
            asm volatile("s_wait_asynccnt 0" ::: "memory");
        }
        __syncthreads();

        // S = Q(16x64) @ K^T(64x32): two key sub-tiles, K-dim split 0..31/32..63
        v8f s0 = {}, s1 = {};
        {
            Frag bk;
            bk.q[0] = *(const uint4*)(Klds + l16 * 64 + grp);
            bk.q[1] = *(const uint4*)(Klds + l16 * 64 + grp + 8);
            s0 = __builtin_amdgcn_wmma_f32_16x16x32_bf16(false, aq0.v, false, bk.v, (short)0, s0, false, false);
            bk.q[0] = *(const uint4*)(Klds + l16 * 64 + 32 + grp);
            bk.q[1] = *(const uint4*)(Klds + l16 * 64 + 32 + grp + 8);
            s0 = __builtin_amdgcn_wmma_f32_16x16x32_bf16(false, aq1.v, false, bk.v, (short)0, s0, false, false);
            bk.q[0] = *(const uint4*)(Klds + (16 + l16) * 64 + grp);
            bk.q[1] = *(const uint4*)(Klds + (16 + l16) * 64 + grp + 8);
            s1 = __builtin_amdgcn_wmma_f32_16x16x32_bf16(false, aq0.v, false, bk.v, (short)0, s1, false, false);
            bk.q[0] = *(const uint4*)(Klds + (16 + l16) * 64 + 32 + grp);
            bk.q[1] = *(const uint4*)(Klds + (16 + l16) * 64 + 32 + grp + 8);
            s1 = __builtin_amdgcn_wmma_f32_16x16x32_bf16(false, aq1.v, false, bk.v, (short)0, s1, false, false);
        }

        // Online softmax over the 32-key chunk; rows live across 16-lane halves.
        #pragma unroll
        for (int r = 0; r < 8; ++r) {
            float v0 = s0[r], v1 = s1[r];
            float rmax = fmaxf(v0, v1);
            #pragma unroll
            for (int sh = 1; sh < 16; sh <<= 1) rmax = fmaxf(rmax, __shfl_xor(rmax, sh, 16));
            const float newm  = fmaxf(mi[r], rmax);
            const float alpha = __expf(mi[r] - newm);
            const float p0 = __expf(v0 - newm);
            const float p1 = __expf(v1 - newm);
            float rsum = p0 + p1;
            #pragma unroll
            for (int sh = 1; sh < 16; sh <<= 1) rsum += __shfl_xor(rsum, sh, 16);
            li[r] = li[r] * alpha + rsum;
            mi[r] = newm;
            acc0[r] *= alpha; acc1[r] *= alpha; acc2[r] *= alpha; acc3[r] *= alpha;
            const int m = r + (hi ? 8 : 0);
            Plds[wid][m * 32 + l16]      = (__bf16)p0;
            Plds[wid][m * 32 + 16 + l16] = (__bf16)p1;
        }
        __syncthreads();   // order P stores before A-fragment reads

        // O += P(16x32) @ V(32x64): one A fragment, four d-tiles of B from VT.
        Frag ap;
        {
            const __bf16* prow = &Plds[wid][l16 * 32];
            const int off = hi ? 16 : 0;
            ap.q[0] = *(const uint4*)(prow + off);
            ap.q[1] = *(const uint4*)(prow + off + 8);
        }
        {
            Frag bv;
            bv.q[0] = *(const uint4*)(VTlds + (0 * 16 + l16) * 32 + grp);
            bv.q[1] = *(const uint4*)(VTlds + (0 * 16 + l16) * 32 + grp + 8);
            acc0 = __builtin_amdgcn_wmma_f32_16x16x32_bf16(false, ap.v, false, bv.v, (short)0, acc0, false, false);
            bv.q[0] = *(const uint4*)(VTlds + (1 * 16 + l16) * 32 + grp);
            bv.q[1] = *(const uint4*)(VTlds + (1 * 16 + l16) * 32 + grp + 8);
            acc1 = __builtin_amdgcn_wmma_f32_16x16x32_bf16(false, ap.v, false, bv.v, (short)0, acc1, false, false);
            bv.q[0] = *(const uint4*)(VTlds + (2 * 16 + l16) * 32 + grp);
            bv.q[1] = *(const uint4*)(VTlds + (2 * 16 + l16) * 32 + grp + 8);
            acc2 = __builtin_amdgcn_wmma_f32_16x16x32_bf16(false, ap.v, false, bv.v, (short)0, acc2, false, false);
            bv.q[0] = *(const uint4*)(VTlds + (3 * 16 + l16) * 32 + grp);
            bv.q[1] = *(const uint4*)(VTlds + (3 * 16 + l16) * 32 + grp + 8);
            acc3 = __builtin_amdgcn_wmma_f32_16x16x32_bf16(false, ap.v, false, bv.v, (short)0, acc3, false, false);
        }
    }

    // Normalize and store: out[b, n, h*64 + d] fp32.
    #pragma unroll
    for (int r = 0; r < 8; ++r) {
        const float inv = 1.0f / li[r];
        const int m = r + (hi ? 8 : 0);
        const size_t base = ((size_t)b * SEQ + (q0 + m)) * DIM + h * HD + l16;
        Out[base +  0] = acc0[r] * inv;
        Out[base + 16] = acc1[r] * inv;
        Out[base + 32] = acc2[r] * inv;
        Out[base + 48] = acc3[r] * inv;
    }
}

extern "C" void kernel_launch(void* const* d_in, const int* in_sizes, int n_in,
                              void* d_out, int out_size, void* d_ws, size_t ws_size,
                              hipStream_t stream) {
    const float* X = (const float*)d_in[0];   // [4,2048,512]
    const float* W = (const float*)d_in[1];   // [512,1536]
    const size_t headElems = (size_t)BATCH * HEADS * SEQ * HD;   // 4M elems each
    __bf16* Q  = (__bf16*)d_ws;
    __bf16* Kp = Q + headElems;
    __bf16* Vp = Kp + headElems;

    qkv_kernel<<<6144, 256, 0, stream>>>(X, W, Q, Kp, Vp);
    attn_kernel<<<512, 256, 0, stream>>>(Q, Kp, Vp, (float*)d_out);
}
